// GGNNRelReason_77129022701589
// MI455X (gfx1250) — matrix operand order repaired and checked by reading, not compiled, not run on hardware
//
#include <hip/hip_runtime.h>
#include <hip/hip_bf16.h>
#include <stdint.h>

typedef __attribute__((ext_vector_type(16))) _Float16 v16h;
typedef __attribute__((ext_vector_type(8)))  _Float16 v8h;
typedef __attribute__((ext_vector_type(8)))  float    v8f;
typedef __attribute__((ext_vector_type(4)))  float    v4f;

// ---------------------------------------------------------------------------
// WMMA GEMM:  C[M x ldc] (f32)  =  A[M x K] (f32, row-major, cvt->f16)
//                                @ Bt^T  where Bt is [ncols][K] f16 row-major.
// One wave computes 16 rows x 64 cols (4 C tiles). 4 waves / block.
// Grid: x = M/64, y = ldc/64 (col group).
// ---------------------------------------------------------------------------
__global__ __launch_bounds__(128) void gemm_f16_wmma(
    const float* __restrict__ A, const _Float16* __restrict__ Bt,
    float* __restrict__ C, int M, int K, int ldc)
{
  const int lane = threadIdx.x & 31;
  const int wave = threadIdx.x >> 5;
  const int row0 = (blockIdx.x * 4 + wave) * 16;
  if (row0 >= M) return;                       // wave-uniform: EXEC stays all-1s
  const int colBase = blockIdx.y * 64;
  const int m  = lane & 15;
  const int hi = lane >> 4;                    // 0 or 1 (lane half)
  // A 16x32 f16 layout: lanes 0-15 hold K = [0..8)+[16..24); lanes 16-31 +8
  const float* arow = A + (size_t)(row0 + m) * K + hi * 8;
  // B 32x16 f16 layout: col n = lane&15; lanes 0-15 hold K = [0..8)+[16..24); lanes 16-31 +8
  const _Float16* b0 = Bt + (size_t)(colBase + m) * K + hi * 8;
  const _Float16* b1 = b0 + (size_t)16 * K;
  const _Float16* b2 = b0 + (size_t)32 * K;
  const _Float16* b3 = b0 + (size_t)48 * K;

  v8f c0{}, c1{}, c2{}, c3{};
#pragma unroll 2
  for (int k = 0; k < K; k += 32) {
    v4f a0 = __builtin_nontemporal_load((const v4f*)(arow + k));
    v4f a1 = __builtin_nontemporal_load((const v4f*)(arow + k + 4));
    v4f a2 = __builtin_nontemporal_load((const v4f*)(arow + k + 16));
    v4f a3 = __builtin_nontemporal_load((const v4f*)(arow + k + 20));
    v16h a;
#pragma unroll
    for (int i = 0; i < 4; ++i) {
      a[i]      = (_Float16)a0[i];
      a[4 + i]  = (_Float16)a1[i];
      a[8 + i]  = (_Float16)a2[i];
      a[12 + i] = (_Float16)a3[i];
    }
    v8h bl, bh; v16h b;
    bl = *(const v8h*)(b0 + k); bh = *(const v8h*)(b0 + k + 16);
#pragma unroll
    for (int i = 0; i < 8; ++i) { b[i] = bl[i]; b[8 + i] = bh[i]; }
    c0 = __builtin_amdgcn_wmma_f32_16x16x32_f16(false, a, false, b, (short)0, c0, false, false);
    bl = *(const v8h*)(b1 + k); bh = *(const v8h*)(b1 + k + 16);
#pragma unroll
    for (int i = 0; i < 8; ++i) { b[i] = bl[i]; b[8 + i] = bh[i]; }
    c1 = __builtin_amdgcn_wmma_f32_16x16x32_f16(false, a, false, b, (short)0, c1, false, false);
    bl = *(const v8h*)(b2 + k); bh = *(const v8h*)(b2 + k + 16);
#pragma unroll
    for (int i = 0; i < 8; ++i) { b[i] = bl[i]; b[8 + i] = bh[i]; }
    c2 = __builtin_amdgcn_wmma_f32_16x16x32_f16(false, a, false, b, (short)0, c2, false, false);
    bl = *(const v8h*)(b3 + k); bh = *(const v8h*)(b3 + k + 16);
#pragma unroll
    for (int i = 0; i < 8; ++i) { b[i] = bl[i]; b[8 + i] = bh[i]; }
    c3 = __builtin_amdgcn_wmma_f32_16x16x32_f16(false, a, false, b, (short)0, c3, false, false);
  }
  // C/D layout: VGPR r -> row M = r + 8*hi ; N = lane&15 within each 16-col tile
  float* crow = C + (size_t)row0 * ldc + colBase + m;
#pragma unroll
  for (int r = 0; r < 8; ++r) {
    float* p = crow + (size_t)(r + hi * 8) * ldc;
    p[0] = c0[r]; p[16] = c1[r]; p[32] = c2[r]; p[48] = c3[r];
  }
}

// ---------------- small prep kernels (scalar fp32, all trivial cost) -------

// Bt3[n][j] = W_head[1024+j][n]  (f16, n<51 else 0) ; 64 x 512
__global__ void build_bt_rel(const float* __restrict__ W_head, _Float16* __restrict__ Bt3) {
  int t = blockIdx.x * blockDim.x + threadIdx.x;
  if (t >= 64 * 512) return;
  int n = t >> 9, j = t & 511;
  float v = (n < 51) ? W_head[(size_t)(1024 + j) * 51 + n] : 0.0f;
  Bt3[(size_t)n * 512 + j] = (_Float16)v;
}

// Ut[n][j]: n<64 -> Wh1+Wv1 col n ; n>=64 -> Wh2+Wv2 col n-64 ; 128 x 512
__global__ void build_ut(const float* __restrict__ W_head, const float* __restrict__ W_v,
                         _Float16* __restrict__ Ut) {
  int t = blockIdx.x * blockDim.x + threadIdx.x;
  if (t >= 128 * 512) return;
  int n = t >> 9, j = t & 511;
  int c = n & 63, half = n >> 6;
  float v = 0.0f;
  if (c < 51) {
    size_t r = (size_t)(half * 512 + j);
    v = W_head[r * 51 + c] + W_v[r * 51 + c];
  }
  Ut[(size_t)n * 512 + j] = (_Float16)v;
}

// dst[n][k] = (f16) src[k][n]   src: M x ld f32 -> dst: ld x M f16
__global__ void transpose_f32_to_f16(const float* __restrict__ src, _Float16* __restrict__ dst,
                                     int M, int ld) {
  int t = blockIdx.x * blockDim.x + threadIdx.x;
  if (t >= M * ld) return;
  int k = t / ld, n = t % ld;
  dst[(size_t)n * M + k] = (_Float16)src[(size_t)k * ld + n];
}

// obj_preds[i] = argmax(logits[i,1:]) + 1  (first max wins)
__global__ void argmax_kernel(const float* __restrict__ logits, int* __restrict__ preds,
                              int N, int C) {
  int i = blockIdx.x * blockDim.x + threadIdx.x;
  if (i >= N) return;
  const float* row = logits + (size_t)i * C;
  float best = row[1]; int bi = 1;
  for (int c = 2; c < C; ++c) { float v = row[c]; if (v > best) { best = v; bi = c; } }
  preds[i] = bi;
}

// T1[cls][n] = Emb[cls]@W_head[1558:1622][:,n], T2 with 1622:1686
__global__ void build_tables(const float* __restrict__ Emb, const float* __restrict__ W_head,
                             float* __restrict__ T1, float* __restrict__ T2) {
  int t = blockIdx.x * blockDim.x + threadIdx.x;
  if (t >= 151 * 51) return;
  int cls = t / 51, n = t % 51;
  float a1 = 0.0f, a2 = 0.0f;
  for (int j = 0; j < 64; ++j) {
    float e = Emb[(size_t)cls * 64 + j];
    a1 += e * W_head[(size_t)(1558 + j) * 51 + n];
    a2 += e * W_head[(size_t)(1622 + j) * 51 + n];
  }
  T1[t] = a1; T2[t] = a2;
}

// biasc[n] = b_head+b_v + b_rel@Wh3 + b_obj@(Wh1+Wv1) + b_obj@(Wh2+Wv2)
__global__ void build_bias(const float* __restrict__ b_head, const float* __restrict__ b_v,
                           const float* __restrict__ b_rel, const float* __restrict__ b_obj,
                           const float* __restrict__ W_head, const float* __restrict__ W_v,
                           float* __restrict__ biasc) {
  int n = threadIdx.x;
  if (n >= 51) return;
  float acc = b_head[n] + b_v[n];
  for (int j = 0; j < 512; ++j) {
    acc += b_rel[j] * W_head[(size_t)(1024 + j) * 51 + n];
    acc += b_obj[j] * (W_head[(size_t)j * 51 + n] + W_v[(size_t)j * 51 + n]);
    acc += b_obj[j] * (W_head[(size_t)(512 + j) * 51 + n] + W_v[(size_t)(512 + j) * 51 + n]);
  }
  biasc[n] = acc;
}

// out[e][c] = R[e][c] + G[s][c] + G[o][64+c] + boxfeat.Wh4 + T1[ps][c] + T2[po][c] + biasc[c]
__global__ void epilogue_kernel(const float* __restrict__ R, const float* __restrict__ G,
                                const float* __restrict__ T1, const float* __restrict__ T2,
                                const float* __restrict__ biasc, const int* __restrict__ rel_inds,
                                const int* __restrict__ preds, const float* __restrict__ bboxes,
                                const float* __restrict__ W_head, float* __restrict__ out, int E) {
  int e = blockIdx.x * blockDim.x + threadIdx.x;
  if (e >= E) return;
  int s = rel_inds[e * 3 + 1];
  int o = rel_inds[e * 3 + 2];
  float bs0 = bboxes[s*4+0], bs1 = bboxes[s*4+1], bs2 = bboxes[s*4+2], bs3 = bboxes[s*4+3];
  float bo0 = bboxes[o*4+0], bo1 = bboxes[o*4+1], bo2 = bboxes[o*4+2], bo3 = bboxes[o*4+3];
  float p0 = fminf(bs0, bo0), p1 = fminf(bs1, bo1), p2 = fmaxf(bs2, bo2), p3 = fmaxf(bs3, bo3);
  // center form (cx, cy, half-w, half-h)
  float scx = (bs0+bs2)*0.5f, scy = (bs1+bs3)*0.5f, sw = (bs2-bs0)*0.5f, sh = (bs3-bs1)*0.5f;
  float ocx = (bo0+bo2)*0.5f, ocy = (bo1+bo3)*0.5f, ow = (bo2-bo0)*0.5f, oh = (bo3-bo1)*0.5f;
  float pcx = (p0+p2)*0.5f,   pcy = (p1+p3)*0.5f,   pw = (p2-p0)*0.5f,   ph = (p3-p1)*0.5f;
  const float inv = 1.0f / 592.0f;
  float f[22];
  f[0]=(scx-ocx)/ow; f[1]=(scy-ocy)/oh; f[2]=__logf(sw/ow); f[3]=__logf(sh*oh);
  f[4]=(scx-pcx)/pw; f[5]=(scy-pcy)/ph; f[6]=__logf(sw/pw); f[7]=__logf(sh*ph);
  f[8]=(pcx-ocx)/ow; f[9]=(pcy-ocy)/oh; f[10]=__logf(pw/ow); f[11]=__logf(ph*oh);
  f[12]=bs0*inv; f[13]=bs1*inv; f[14]=(bs0+bs2)*inv; f[15]=(bs1+bs3)*inv; f[16]=bs2*bs3*inv*inv;
  f[17]=bo0*inv; f[18]=bo1*inv; f[19]=(bo0+bo2)*inv; f[20]=(bo1+bo3)*inv; f[21]=bo2*bo3*inv*inv;
  int ps = preds[s], po = preds[o];
  const float* Re = R + (size_t)e * 64;
  const float* Gs = G + (size_t)s * 128;
  const float* Go = G + (size_t)o * 128 + 64;
  const float* Ts = T1 + (size_t)ps * 51;
  const float* To = T2 + (size_t)po * 51;
  float* oe = out + (size_t)e * 51;
  for (int c = 0; c < 51; ++c) {
    float acc = Re[c] + Gs[c] + Go[c] + Ts[c] + To[c] + biasc[c];
#pragma unroll
    for (int j = 0; j < 22; ++j) acc += f[j] * W_head[(size_t)(1536 + j) * 51 + c];
    oe[c] = acc;
  }
}

// ---------------------------------------------------------------------------
extern "C" void kernel_launch(void* const* d_in, const int* in_sizes, int n_in,
                              void* d_out, int out_size, void* d_ws, size_t ws_size,
                              hipStream_t stream) {
  const float* obj_fmaps  = (const float*)d_in[0];
  const float* obj_logits = (const float*)d_in[1];
  const float* vr         = (const float*)d_in[2];
  const float* bboxes     = (const float*)d_in[3];
  const float* W_obj      = (const float*)d_in[4];
  const float* b_obj      = (const float*)d_in[5];
  const float* W_rel      = (const float*)d_in[6];
  const float* b_rel      = (const float*)d_in[7];
  const float* Emb        = (const float*)d_in[8];
  const float* W_head     = (const float*)d_in[9];
  const float* b_head     = (const float*)d_in[10];
  const float* W_v        = (const float*)d_in[11];
  const float* b_v        = (const float*)d_in[12];
  const int*   rel_inds   = (const int*)d_in[13];
  (void)in_sizes; (void)n_in; (void)out_size; (void)ws_size;

  const int N = 1024, E = 32768, K = 4096, NOBJ = 151;

  char* ws = (char*)d_ws;
  size_t off = 0;
  auto carve = [&](size_t bytes) -> void* {
    void* p = ws + off;
    off = (off + bytes + 255) & ~(size_t)255;
    return p;
  };
  _Float16* Pt    = (_Float16*)carve((size_t)64  * K * 2);   // P^T  f16 [64][4096]
  _Float16* Qt    = (_Float16*)carve((size_t)128 * K * 2);   // Q^T  f16 [128][4096]
  float*    P32   = (float*)   carve((size_t)K * 64 * 4);    // W_rel@Wh3, 4096x64
  float*    Q32   = (float*)   carve((size_t)K * 128 * 4);   // W_obj@[U1|U2], 4096x128
  _Float16* Bt3   = (_Float16*)carve((size_t)64  * 512 * 2); // Wh3^T f16
  _Float16* Ut    = (_Float16*)carve((size_t)128 * 512 * 2); // [U1|U2]^T f16
  float*    R     = (float*)   carve((size_t)E * 64 * 4);    // vr@P, 32768x64
  float*    G     = (float*)   carve((size_t)N * 128 * 4);   // obj_fmaps@Q, 1024x128
  float*    T1    = (float*)   carve((size_t)151 * 51 * 4);
  float*    T2    = (float*)   carve((size_t)151 * 51 * 4);
  float*    biasc = (float*)   carve(64 * 4);
  int*      preds = (int*)     carve((size_t)N * 4);

  // 1) fold head weights
  build_bt_rel<<<(64 * 512) / 256, 256, 0, stream>>>(W_head, Bt3);
  build_ut<<<(128 * 512) / 256, 256, 0, stream>>>(W_head, W_v, Ut);
  // 2) P = W_rel @ Wh3 ; Q = W_obj @ [U1|U2]   (WMMA, K=512)
  gemm_f16_wmma<<<dim3(K / 64, 1), 128, 0, stream>>>(W_rel, Bt3, P32, K, 512, 64);
  gemm_f16_wmma<<<dim3(K / 64, 2), 128, 0, stream>>>(W_obj, Ut, Q32, K, 512, 128);
  transpose_f32_to_f16<<<(K * 64) / 256, 256, 0, stream>>>(P32, Pt, K, 64);
  transpose_f32_to_f16<<<(K * 128) / 256, 256, 0, stream>>>(Q32, Qt, K, 128);
  // 3) main HBM-bound GEMMs (WMMA, K=4096): R = vr@P, G = obj_fmaps@Q
  gemm_f16_wmma<<<dim3(E / 64, 1), 128, 0, stream>>>(vr, Pt, R, E, K, 64);
  gemm_f16_wmma<<<dim3(N / 64, 2), 128, 0, stream>>>(obj_fmaps, Qt, G, N, K, 128);
  // 4) class tables, argmax, fused bias
  argmax_kernel<<<(N + 255) / 256, 256, 0, stream>>>(obj_logits, preds, N, NOBJ);
  build_tables<<<(151 * 51 + 255) / 256, 256, 0, stream>>>(Emb, W_head, T1, T2);
  build_bias<<<1, 64, 0, stream>>>(b_head, b_v, b_rel, b_obj, W_head, W_v, biasc);
  // 5) per-edge epilogue: gathers + box features + final sum
  epilogue_kernel<<<(E + 127) / 128, 128, 0, stream>>>(R, G, T1, T2, biasc, rel_inds,
                                                       preds, bboxes, W_head, (float*)d_out, E);
}
